// Attention_15118284882388
// MI455X (gfx1250) — matrix operand order repaired and checked by reading, not compiled, not run on hardware
//
#include <hip/hip_runtime.h>

typedef unsigned short u16;
typedef __attribute__((ext_vector_type(16))) __bf16 v16bf;
typedef __attribute__((ext_vector_type(8)))  __bf16 v8bf;
typedef __attribute__((ext_vector_type(8)))  float  v8f;
typedef unsigned int u32x4 __attribute__((ext_vector_type(4)));
typedef int          i32x4 __attribute__((ext_vector_type(4)));
typedef int          i32x8 __attribute__((ext_vector_type(8)));
typedef __attribute__((address_space(3))) u16 lds_u16;

#if defined(__gfx1250__) && __has_builtin(__builtin_amdgcn_tensor_load_to_lds)
#define HAVE_TDM 1
#else
#define HAVE_TDM 0
#endif

#define DEV __device__ __forceinline__

constexpr int Bsz  = 16;
constexpr int Cdim = 512;
constexpr int Seq  = 1024;   // 32*32
constexpr int NH   = 8;
constexpr int HD   = 64;
constexpr int C3   = 3 * Cdim;
constexpr int Mrows = Bsz * Seq;            // 16384
constexpr size_t QKVPART = (size_t)Bsz * NH * Seq * HD; // 8388608 elems per {Q,K,V}

// ---------- helpers ----------

DEV u16 f2bf(float x) {                       // f32 -> bf16 RNE
  unsigned u = __float_as_uint(x);
  u += 0x7FFFu + ((u >> 16) & 1u);
  return (u16)(u >> 16);
}

// A-matrix 16x32 bf16 fragment (ISA 7.12.2): lanes 0-15 row M=lane, K {0..7,16..23};
// lanes 16-31 same row, K {8..15,24..31}. `row` points at (row start + k0).
DEV v16bf load_a16x32(const u16* row, int lane) {
  int hi = (lane >> 4) & 1;
  v16bf r;
  ((v8bf*)&r)[0] = *(const v8bf*)(row + hi * 8);
  ((v8bf*)&r)[1] = *(const v8bf*)(row + hi * 8 + 16);
  return r;
}

// B-matrix 32x16 bf16 fragment: lane holds column N=lane&15; lanes 0-15 K {0..7,16..23},
// lanes 16-31 K {8..15,24..31} (same k order as A). Source row-major, so
// a "column of B" is a contiguous row of the source (W / K / V^T).
DEV v16bf load_b32x16(const u16* row, int lane) {
  int hi = (lane >> 4) & 1;
  v16bf r;
  ((v8bf*)&r)[0] = *(const v8bf*)(row + hi * 8);
  ((v8bf*)&r)[1] = *(const v8bf*)(row + hi * 8 + 16);
  return r;
}

DEV v8f wmma_bf(v16bf a, v16bf b, v8f c) {
  return __builtin_amdgcn_wmma_f32_16x16x32_bf16(false, a, false, b, (short)0, c,
                                                 false, false);
}

// reductions over each 16-lane half of the wave (rows of D layout)
DEV float half16_max(float x) {
#pragma unroll
  for (int m = 1; m < 16; m <<= 1) x = fmaxf(x, __shfl_xor(x, m, 32));
  return x;
}
DEV float half16_sum(float x) {
#pragma unroll
  for (int m = 1; m < 16; m <<= 1) x += __shfl_xor(x, m, 32);
  return x;
}

// ---------- kernel 1: convert + layout ----------
// x: (B,C,32,32) f32 -> Xb: (B,S,C) bf16 ; weights f32 -> bf16
__global__ void __launch_bounds__(256)
convert_kernel(const float* __restrict__ x, const float* __restrict__ wq,
               const float* __restrict__ wo, u16* __restrict__ Xb,
               u16* __restrict__ Wqb, u16* __restrict__ Wob) {
  const size_t stride = (size_t)gridDim.x * blockDim.x;
  const size_t tid = (size_t)blockIdx.x * blockDim.x + threadIdx.x;
  const size_t XN = (size_t)Bsz * Cdim * Seq;
  for (size_t i = tid; i < XN; i += stride) {
    size_t b = i / ((size_t)Cdim * Seq);
    size_t c = (i / Seq) % Cdim;
    size_t s = i % Seq;
    Xb[(b * Seq + s) * Cdim + c] = f2bf(x[i]);   // coalesced read, scattered bf16 write
  }
  for (size_t i = tid; i < (size_t)C3 * Cdim; i += stride) Wqb[i] = f2bf(wq[i]);
  for (size_t i = tid; i < (size_t)Cdim * Cdim; i += stride) Wob[i] = f2bf(wo[i]);
}

// ---------- kernel 2: QKV projection ----------
// Y = Xb @ Wq^T  (16384x512 @ 512x1536), scatter into Q/K/V (B,NH,S,HD) bf16
__global__ void __launch_bounds__(256)
qkv_gemm(const u16* __restrict__ Xb, const u16* __restrict__ Wq,
         u16* __restrict__ QKV) {
  const int lane = threadIdx.x & 31;
  const int w = threadIdx.x >> 5;
  const int wg = blockIdx.x * 8 + w;
  const int NT = C3 / 64;                    // 24 n-tiles
  const int mt = (wg / NT) * 64;
  const int nt = (wg % NT) * 64;

  v8f acc[4][4];
#pragma unroll
  for (int i = 0; i < 4; i++)
#pragma unroll
    for (int j = 0; j < 4; j++) acc[i][j] = v8f{0, 0, 0, 0, 0, 0, 0, 0};

  const u16* arow[4];
  const u16* brow[4];
#pragma unroll
  for (int i = 0; i < 4; i++) arow[i] = Xb + (size_t)(mt + i * 16 + (lane & 15)) * Cdim;
#pragma unroll
  for (int j = 0; j < 4; j++) brow[j] = Wq + (size_t)(nt + j * 16 + (lane & 15)) * Cdim;

  for (int k = 0; k < Cdim; k += 32) {
    if (k + 32 < Cdim) {
      __builtin_prefetch(arow[0] + k + 32, 0, 1);
      __builtin_prefetch(brow[0] + k + 32, 0, 1);
    }
    v16bf af[4], bf4[4];
#pragma unroll
    for (int i = 0; i < 4; i++) af[i] = load_a16x32(arow[i] + k, lane);
#pragma unroll
    for (int j = 0; j < 4; j++) bf4[j] = load_b32x16(brow[j] + k, lane);
#pragma unroll
    for (int i = 0; i < 4; i++)
#pragma unroll
      for (int j = 0; j < 4; j++) acc[i][j] = wmma_bf(af[i], bf4[j], acc[i][j]);
  }

  // epilogue: d = nt + j*16 + (lane&15); part/head constant per 64-wide tile
  const int b = mt >> 10;
  const int s0 = mt & (Seq - 1);
  const int part = nt >> 9;
  const int head = (nt >> 6) & (NH - 1);
  u16* dst = QKV + (size_t)part * QKVPART + ((size_t)(b * NH + head)) * Seq * HD;
  const int ncol = lane & 15;
  const int rowoff = (lane >> 4) * 8;
#pragma unroll
  for (int i = 0; i < 4; i++)
#pragma unroll
    for (int j = 0; j < 4; j++) {
      const int e = j * 16 + ncol;
#pragma unroll
      for (int r = 0; r < 8; r++) {
        const int s = s0 + i * 16 + rowoff + r;
        dst[(size_t)s * HD + e] = f2bf(acc[i][j][r]);
      }
    }
}

// ---------- kernel 3: flash attention ----------
// one block = (b, head, 128 Q rows); 8 waves x 16 rows each
__global__ void __launch_bounds__(256)
attn_kernel(const u16* __restrict__ QKV, u16* __restrict__ Obuf) {
  constexpr int VTS = 144;  // padded V^T row stride (16B aligned, conflict-breaking)
  __shared__ __align__(16) u16 kbuf[128 * 64];
  __shared__ __align__(16) u16 vt[64 * VTS];
  __shared__ __align__(16) u16 pbuf[8][16 * 32];

  const int tid = threadIdx.x;
  const int lane = tid & 31;
  const int w = tid >> 5;
  const int bid = blockIdx.x;
  const int b = bid >> 6;
  const int h = (bid >> 3) & 7;
  const int q0 = (bid & 7) * 128;

  const u16* Qh = QKV + 0 * QKVPART + ((size_t)(b * NH + h)) * Seq * HD;
  const u16* Kh = QKV + 1 * QKVPART + ((size_t)(b * NH + h)) * Seq * HD;
  const u16* Vh = QKV + 2 * QKVPART + ((size_t)(b * NH + h)) * Seq * HD;

  const int ncol = lane & 15;
  const int rowoff = (lane >> 4) * 8;

  // Q fragments live in registers for the whole kernel (16 rows x 64)
  const int qrow = q0 + w * 16 + (lane & 15);
  v16bf aq[2];
  aq[0] = load_a16x32(Qh + (size_t)qrow * HD + 0, lane);
  aq[1] = load_a16x32(Qh + (size_t)qrow * HD + 32, lane);

  v8f ot[4];
#pragma unroll
  for (int nb = 0; nb < 4; nb++) ot[nb] = v8f{0, 0, 0, 0, 0, 0, 0, 0};
  float rmax[8], rsum[8];
#pragma unroll
  for (int r = 0; r < 8; r++) { rmax[r] = -1e30f; rsum[r] = 0.0f; }

#if HAVE_TDM
  // LDS byte offset of kbuf for the TDM descriptor
  const unsigned kbuf_lds = (unsigned)(unsigned long long)(lds_u16*)kbuf;
#endif

  for (int kc = 0; kc < Seq; kc += 128) {
#if HAVE_TDM
    // --- Tensor Data Mover: stage K chunk (128 keys x 64 feats, bf16) to LDS ---
    // D# per ISA ch.8: 2D tensor dim0=HD (stride HD), dim1=Seq; tile 64x128;
    // data_size=2B; groups 2/3 disabled (zeros); issued by wave 0 only
    // (EXEC ignored by TENSOR ops), completion via TENSORcnt.
    if (w == 0) {
      unsigned long long ga = (unsigned long long)(Kh + (size_t)kc * HD);
      u32x4 g0;
      g0[0] = 1u;                                     // count=1, user descriptor
      g0[1] = kbuf_lds;                               // lds_addr
      g0[2] = (unsigned)ga;                           // global_addr[31:0]
      g0[3] = (unsigned)((ga >> 32) & 0x01FFFFFFull)  // global_addr[56:32]
              | 0x80000000u;                          // type=2 ("image")
      i32x8 g1;
      g1[0] = 0x00010000;                             // data_size=1 (2 bytes)
      g1[1] = (int)((unsigned)HD << 16);              // tensor_dim0=64  (bits 79:48)
      g1[2] = (int)((unsigned)Seq << 16);             // tensor_dim1=1024(bits 111:80)
      g1[3] = (int)((unsigned)HD << 16);              // tile_dim0=64    (bits 127:112)
      g1[4] = 128;                                    // tile_dim1=128, tile_dim2=0
      g1[5] = HD;                                     // tensor_dim0_stride=64
      g1[6] = 0;
      g1[7] = 0;
      i32x4 g2 = {0, 0, 0, 0};
      i32x4 g3 = {0, 0, 0, 0};
#if __clang_major__ >= 23
      i32x8 g4 = {0, 0, 0, 0, 0, 0, 0, 0};
      __builtin_amdgcn_tensor_load_to_lds(g0, g1, g2, g3, g4, 0);
#else
      __builtin_amdgcn_tensor_load_to_lds(g0, g1, g2, g3, 0);
#endif
      __builtin_amdgcn_s_wait_tensorcnt(0);
    }
#else
    // fallback: cooperative vector copy of K chunk
    {
      const uint4* src = (const uint4*)(Kh + (size_t)kc * HD);
      uint4* dst = (uint4*)kbuf;
#pragma unroll
      for (int i = 0; i < 4; i++) dst[tid + i * 256] = src[tid + i * 256];
    }
#endif
    // stage V chunk transposed: vt[feature][key_in_chunk] (TDM cannot transpose)
#pragma unroll
    for (int i = 0; i < 4; i++) {
      const int task = tid + i * 256;      // 0..1023
      const int s = task >> 3;             // key 0..127
      const int ch = task & 7;             // 8-feature chunk
      uint4 val = *(const uint4*)(Vh + (size_t)(kc + s) * HD + ch * 8);
      const u16* u = (const u16*)&val;
#pragma unroll
      for (int j = 0; j < 8; j++) vt[(ch * 8 + j) * VTS + s] = u[j];
    }
    __syncthreads();

    for (int kk = 0; kk < 4; kk++) {       // 32 keys per iteration
      v8f s0 = v8f{0, 0, 0, 0, 0, 0, 0, 0};
      v8f s1 = v8f{0, 0, 0, 0, 0, 0, 0, 0};
#pragma unroll
      for (int ks = 0; ks < 2; ks++) {     // K-dim 64 in two steps
        v16bf b0 = load_b32x16(&kbuf[(kk * 32 + (lane & 15)) * HD] + ks * 32, lane);
        v16bf b1 = load_b32x16(&kbuf[(kk * 32 + 16 + (lane & 15)) * HD] + ks * 32, lane);
        s0 = wmma_bf(aq[ks], b0, s0);
        s1 = wmma_bf(aq[ks], b1, s1);
      }
      // online softmax over these 32 columns
      float p0[8], p1[8], fac[8];
#pragma unroll
      for (int r = 0; r < 8; r++) {
        const float a = s0[r] * 0.125f;    // 1/sqrt(64)
        const float c = s1[r] * 0.125f;
        const float tm = half16_max(fmaxf(a, c));
        const float nm = fmaxf(rmax[r], tm);
        fac[r] = __expf(rmax[r] - nm);
        p0[r] = __expf(a - nm);
        p1[r] = __expf(c - nm);
        const float ts = half16_sum(p0[r] + p1[r]);
        rsum[r] = rsum[r] * fac[r] + ts;
        rmax[r] = nm;
      }
#pragma unroll
      for (int nb = 0; nb < 4; nb++)
#pragma unroll
        for (int r = 0; r < 8; r++) ot[nb][r] *= fac[r];

      // D-layout P -> A-layout via per-wave LDS buffer
#pragma unroll
      for (int r = 0; r < 8; r++) {
        pbuf[w][(rowoff + r) * 32 + ncol] = f2bf(p0[r]);
        pbuf[w][(rowoff + r) * 32 + 16 + ncol] = f2bf(p1[r]);
      }
      asm volatile("s_wait_dscnt 0" ::: "memory");  // wave-local RAW on LDS
      v16bf ap = load_a16x32(&pbuf[w][(lane & 15) * 32], lane);
#pragma unroll
      for (int nb = 0; nb < 4; nb++) {
        v16bf bv = load_b32x16(&vt[(nb * 16 + (lane & 15)) * VTS] + kk * 32, lane);
        ot[nb] = wmma_bf(ap, bv, ot[nb]);
      }
    }
    __syncthreads();
  }

  // finalize: O /= rowsum, write (B,S,C) bf16
  float inv[8];
#pragma unroll
  for (int r = 0; r < 8; r++) inv[r] = 1.0f / rsum[r];
  const int srow0 = q0 + w * 16 + rowoff;
#pragma unroll
  for (int nb = 0; nb < 4; nb++)
#pragma unroll
    for (int r = 0; r < 8; r++) {
      const int s = srow0 + r;
      const int c = h * HD + nb * 16 + ncol;
      Obuf[((size_t)(b * Seq + s)) * Cdim + c] = f2bf(ot[nb][r] * inv[r]);
    }
}

// ---------- kernel 4: output projection ----------
// out = Obuf @ Wo^T, stored NCHW f32
__global__ void __launch_bounds__(256)
out_gemm(const u16* __restrict__ Ob, const u16* __restrict__ Wo,
         float* __restrict__ out) {
  const int lane = threadIdx.x & 31;
  const int w = threadIdx.x >> 5;
  const int wg = blockIdx.x * 8 + w;
  const int NT = Cdim / 64;                 // 8 n-tiles
  const int mt = (wg / NT) * 64;
  const int nt = (wg % NT) * 64;

  v8f acc[4][4];
#pragma unroll
  for (int i = 0; i < 4; i++)
#pragma unroll
    for (int j = 0; j < 4; j++) acc[i][j] = v8f{0, 0, 0, 0, 0, 0, 0, 0};

  const u16* arow[4];
  const u16* brow[4];
#pragma unroll
  for (int i = 0; i < 4; i++) arow[i] = Ob + (size_t)(mt + i * 16 + (lane & 15)) * Cdim;
#pragma unroll
  for (int j = 0; j < 4; j++) brow[j] = Wo + (size_t)(nt + j * 16 + (lane & 15)) * Cdim;

  for (int k = 0; k < Cdim; k += 32) {
    if (k + 32 < Cdim) {
      __builtin_prefetch(arow[0] + k + 32, 0, 1);
      __builtin_prefetch(brow[0] + k + 32, 0, 1);
    }
    v16bf af[4], bf4[4];
#pragma unroll
    for (int i = 0; i < 4; i++) af[i] = load_a16x32(arow[i] + k, lane);
#pragma unroll
    for (int j = 0; j < 4; j++) bf4[j] = load_b32x16(brow[j] + k, lane);
#pragma unroll
    for (int i = 0; i < 4; i++)
#pragma unroll
      for (int j = 0; j < 4; j++) acc[i][j] = wmma_bf(af[i], bf4[j], acc[i][j]);
  }

  // epilogue: out[b][n][s] ; lane owns column n -> contiguous s => float4 stores
  const int b = mt >> 10;
  const int s0 = mt & (Seq - 1);
  const int ncol = lane & 15;
  const int rowoff = (lane >> 4) * 8;
#pragma unroll
  for (int i = 0; i < 4; i++)
#pragma unroll
    for (int j = 0; j < 4; j++) {
      const int n = nt + j * 16 + ncol;
      const int s = s0 + i * 16 + rowoff;
      float* p = out + ((size_t)(b * Cdim + n)) * Seq + s;
      float4 lo = make_float4(acc[i][j][0], acc[i][j][1], acc[i][j][2], acc[i][j][3]);
      float4 hi = make_float4(acc[i][j][4], acc[i][j][5], acc[i][j][6], acc[i][j][7]);
      *(float4*)p = lo;
      *(float4*)(p + 4) = hi;
    }
}

// ---------- host ----------
extern "C" void kernel_launch(void* const* d_in, const int* in_sizes, int n_in,
                              void* d_out, int out_size, void* d_ws, size_t ws_size,
                              hipStream_t stream) {
  const float* x  = (const float*)d_in[0];   // (16,512,32,32)
  const float* wq = (const float*)d_in[1];   // (1536,512)
  const float* wo = (const float*)d_in[2];   // (512,512)
  float* out = (float*)d_out;                // (16,512,32,32)

  u16* ws = (u16*)d_ws;
  u16* Xb  = ws;                                         // 8,388,608 elems (reused as O)
  u16* Wqb = Xb + (size_t)Mrows * Cdim;                  // 786,432
  u16* Wob = Wqb + (size_t)C3 * Cdim;                    // 262,144
  u16* QKV = Wob + (size_t)Cdim * Cdim;                  // 3 * 8,388,608
  u16* Ob  = Xb;                                         // alias: Xb dead after qkv_gemm

  convert_kernel<<<4096, 256, 0, stream>>>(x, wq, wo, Xb, Wqb, Wob);

  const int qkv_blocks = (Mrows / 64) * (C3 / 64) / 8;   // 768
  qkv_gemm<<<qkv_blocks, 256, 0, stream>>>(Xb, Wqb, QKV);

  const int attn_blocks = Bsz * NH * (Seq / 128);        // 1024
  attn_kernel<<<attn_blocks, 256, 0, stream>>>(QKV, Ob);

  const int out_blocks = (Mrows / 64) * (Cdim / 64) / 8; // 256
  out_gemm<<<out_blocks, 256, 0, stream>>>(Ob, Wob, out);
}
